// GroupedQueryAttention_62311385530775
// MI455X (gfx1250) — matrix operand order repaired and checked by reading, not compiled, not run on hardware
//
#include <hip/hip_runtime.h>
#include <hip/hip_bf16.h>

typedef __attribute__((ext_vector_type(16))) _Float16 v16h;
typedef __attribute__((ext_vector_type(8)))  float    v8f;

union V16 {
    v16h   v;
    float4 f4[2];
    _Float16 h[16];
};

#define WMMA_F16(A, B, C) \
    __builtin_amdgcn_wmma_f32_16x16x32_f16(false, (A), false, (B), (short)0, (C), false, false)

#define LOAD_F4(dst, ptr) (dst) = *(const float4*)(ptr)

// ---------------------------------------------------------------------------
// Generic tiled GEMM:  C[M,N] = A[M,K] @ W[K,N]
//   A: fp32 or f16 (template), W: fp32 (converted to f16 on load), C: f16 or fp32
//   Block = 128 threads (4 waves), macro-tile 128x64, K-step 32.
//   Each wave computes a 32x64 strip: 8 WMMAs per K-step, B-fragments reused
//   across the two 16-row sub-tiles. Bt stored TRANSPOSED so B-fragment
//   gathers are contiguous ds_load_b128.
// ---------------------------------------------------------------------------
template <bool A_HALF, bool OUT_FLOAT>
__global__ __launch_bounds__(128) void gemm_wmma(const void* __restrict__ Aptr,
                                                 const float* __restrict__ W,
                                                 void* __restrict__ Cptr,
                                                 int M, int N, int K) {
    __shared__ _Float16 As[128][40];     // [row][k]   row stride 80B (16B-mult)
    __shared__ _Float16 Bt[64][40];      // [col][k]   transposed B tile

    const int t    = threadIdx.x;
    const int lane = t & 31;
    const int wave = t >> 5;
    const int m    = lane & 15;          // row (A-frag) / col (B-frag) id
    const int h    = lane >> 4;          // lane half
    const int row0 = blockIdx.x * 128;
    const int col0 = blockIdx.y * 64;

    v8f acc[8] = {v8f{}, v8f{}, v8f{}, v8f{}, v8f{}, v8f{}, v8f{}, v8f{}};

    for (int k0 = 0; k0 < K; k0 += 32) {
        __syncthreads();
        // --- fill As: 128 rows x 32 k; thread t -> row t, all 32 cols ---
        {
            V16 u0, u1;
            if constexpr (A_HALF) {
                const _Float16* src = (const _Float16*)Aptr + (size_t)(row0 + t) * K + k0;
                LOAD_F4(u0.f4[0], src);      LOAD_F4(u0.f4[1], src + 8);
                LOAD_F4(u1.f4[0], src + 16); LOAD_F4(u1.f4[1], src + 24);
            } else {
                const float* src = (const float*)Aptr + (size_t)(row0 + t) * K + k0;
                float w[32];
#pragma unroll
                for (int i = 0; i < 8; ++i) LOAD_F4(((float4*)w)[i], src + 4 * i);
#pragma unroll
                for (int i = 0; i < 16; ++i) { u0.h[i] = (_Float16)w[i]; u1.h[i] = (_Float16)w[16 + i]; }
            }
            *(float4*)&As[t][0]  = u0.f4[0];
            *(float4*)&As[t][8]  = u0.f4[1];
            *(float4*)&As[t][16] = u1.f4[0];
            *(float4*)&As[t][24] = u1.f4[1];
        }
        // --- fill Bt (transposed): thread t -> k-row t>>2, 16 cols scattered ---
        {
            const int r = t >> 2;            // k index 0..31
            const int c = (t & 3) * 16;      // col base
            const float* src = W + (size_t)(k0 + r) * N + col0 + c;
            float w[16];
#pragma unroll
            for (int i = 0; i < 4; ++i) LOAD_F4(((float4*)w)[i], src + 4 * i);
#pragma unroll
            for (int i = 0; i < 16; ++i) Bt[c + i][r] = (_Float16)w[i];
        }
        __syncthreads();

        // --- two A fragments (rows m and 16+m of the wave's 32-row strip) ---
        V16 a0, a1;
        {
            const _Float16* s0 = &As[wave * 32 + m][0];
            const _Float16* s1 = &As[wave * 32 + 16 + m][0];
            LOAD_F4(a0.f4[0], s0 + 8 * h);
            LOAD_F4(a0.f4[1], s0 + 16 + 8 * h);
            LOAD_F4(a1.f4[0], s1 + 8 * h);
            LOAD_F4(a1.f4[1], s1 + 16 + 8 * h);
        }
        // --- B fragments (lane col = m, K = 8h+e / 16+8h+e, table order in Bt);
        //     each reused for both row sub-tiles ---
#pragma unroll
        for (int nt = 0; nt < 4; ++nt) {
            V16 b;
            const _Float16* src = &Bt[nt * 16 + m][8 * h];
            LOAD_F4(b.f4[0], src);
            LOAD_F4(b.f4[1], src + 16);
            acc[nt]     = WMMA_F16(a0.v, b.v, acc[nt]);
            acc[4 + nt] = WMMA_F16(a1.v, b.v, acc[4 + nt]);
        }
    }

    // --- store C: acc[nt] -> rows 8h+j, acc[4+nt] -> rows 16+8h+j ---
#pragma unroll
    for (int nt = 0; nt < 4; ++nt) {
#pragma unroll
        for (int j = 0; j < 8; ++j) {
            const size_t r0 = (size_t)(row0 + wave * 32 + 8 * h + j);
            const size_t cc = (size_t)(col0 + nt * 16 + m);
            if constexpr (OUT_FLOAT) {
                ((float*)Cptr)[r0 * N + cc]        = acc[nt][j];
                ((float*)Cptr)[(r0 + 16) * N + cc] = acc[4 + nt][j];
            } else {
                ((_Float16*)Cptr)[r0 * N + cc]        = (_Float16)acc[nt][j];
                ((_Float16*)Cptr)[(r0 + 16) * N + cc] = (_Float16)acc[4 + nt][j];
            }
        }
    }
}

// ---------------------------------------------------------------------------
// Flash-style GQA attention with relative-position bias.
//   Block = 4 waves sharing one (batch, head); each wave owns 16 query rows.
//   K loop in chunks of 32 keys; K/V tiles staged in LDS (V transposed).
//   Softmax denominator via extra WMMA against an all-ones B (P @ 1 = rowsum),
//   so only the row-max needs a cross-lane reduction.
// ---------------------------------------------------------------------------
__global__ __launch_bounds__(128) void attn_wmma(const _Float16* __restrict__ qp,
                                                 const _Float16* __restrict__ kp,
                                                 const _Float16* __restrict__ vp,
                                                 const float* __restrict__ rel_table,
                                                 _Float16* __restrict__ ao) {
    constexpr int S = 484;
    constexpr int MAXD = 483;            // MAX_DIST - 1
    const int qt   = blockIdx.x & 7;     // 8 tiles of 64 q rows
    const int bh   = blockIdx.x >> 3;
    const int head = bh & 15;
    const int b    = bh >> 4;
    const int t    = threadIdx.x;
    const int wave = t >> 5;
    const int lane = t & 31;
    const int m    = lane & 15;
    const int h    = lane >> 4;
    const int q0   = qt * 64 + wave * 16;
    const int g    = head >> 2;          // kv head

    const size_t qbase  = (size_t)b * S * 1024 + (size_t)head * 64;
    const size_t kvbase = (size_t)b * S * 256 + (size_t)g * 64;

    __shared__ _Float16 s_k[32][72];        // [key][d]  row-major
    __shared__ _Float16 s_vt[64][40];       // [d][key]  transposed
    __shared__ _Float16 s_p[4][16][40];     // per-wave P tile (16 q x 32 k)

    // --- load Q fragments once (d 0..31 and 32..63): 4 x b128 ---
    V16 qa0, qa1;
    {
        int qr = q0 + m;
        if (qr >= S) qr = S - 1;
        const _Float16* src = qp + qbase + (size_t)qr * 1024;
        LOAD_F4(qa0.f4[0], src + 16 * h);
        LOAD_F4(qa0.f4[1], src + 16 * h + 8);
        LOAD_F4(qa1.f4[0], src + 32 + 16 * h);
        LOAD_F4(qa1.f4[1], src + 32 + 16 * h + 8);
    }
    // all-ones B fragment for the row-sum WMMA
    V16 ones;
#pragma unroll
    for (int e = 0; e < 16; ++e) ones.h[e] = (_Float16)1.0f;

    float mrow[8];
#pragma unroll
    for (int j = 0; j < 8; ++j) mrow[j] = -1e30f;
    v8f acc[4] = {v8f{}, v8f{}, v8f{}, v8f{}};
    v8f accl = {};                           // running row sums (denominator)

    for (int kc = 0; kc < 16; ++kc) {        // ceil(484/32) = 16 chunks
        const int kb = kc * 32;

        // ---- cooperative fill of K tile and transposed V tile ----
        __syncthreads();
        {
            const int key = t >> 2;          // 0..31
            const int d0  = (t & 3) * 16;    // 0,16,32,48
            int kg = kb + key; if (kg >= S) kg = S - 1;
            const size_t base = kvbase + (size_t)kg * 256 + d0;
            V16 uk, uv;
            LOAD_F4(uk.f4[0], kp + base);
            LOAD_F4(uk.f4[1], kp + base + 8);
            *(float4*)&s_k[key][d0]     = uk.f4[0];
            *(float4*)&s_k[key][d0 + 8] = uk.f4[1];
            LOAD_F4(uv.f4[0], vp + base);
            LOAD_F4(uv.f4[1], vp + base + 8);
#pragma unroll
            for (int i = 0; i < 16; ++i) s_vt[d0 + i][key] = uv.h[i];
        }
        __syncthreads();

        // ---- scores: two 16-key halves, each = 2 WMMAs over D=64 ----
        v8f c0 = {}, c1 = {};
        {
            const _Float16* src = &s_k[m][0];          // key local m
            V16 b0, b1;
            LOAD_F4(b0.f4[0], src + 16 * h);
            LOAD_F4(b0.f4[1], src + 16 * h + 8);
            LOAD_F4(b1.f4[0], src + 32 + 16 * h);
            LOAD_F4(b1.f4[1], src + 32 + 16 * h + 8);
            c0 = WMMA_F16(qa0.v, b0.v, c0);
            c0 = WMMA_F16(qa1.v, b1.v, c0);
        }
        {
            const _Float16* src = &s_k[16 + m][0];     // key local 16+m
            V16 b0, b1;
            LOAD_F4(b0.f4[0], src + 16 * h);
            LOAD_F4(b0.f4[1], src + 16 * h + 8);
            LOAD_F4(b1.f4[0], src + 32 + 16 * h);
            LOAD_F4(b1.f4[1], src + 32 + 16 * h + 8);
            c1 = WMMA_F16(qa0.v, b0.v, c1);
            c1 = WMMA_F16(qa1.v, b1.v, c1);
        }

        // ---- scale + rel-bias + mask (lane holds col m of rows 8h+j) ----
        float s0[8], s1[8];
        const int k0i = kb + m, k1i = kb + 16 + m;
#pragma unroll
        for (int j = 0; j < 8; ++j) {
            const int qr = q0 + 8 * h + j;
            int d0 = k0i - qr; d0 = d0 < -MAXD ? -MAXD : (d0 > MAXD ? MAXD : d0);
            int d1 = k1i - qr; d1 = d1 < -MAXD ? -MAXD : (d1 > MAXD ? MAXD : d1);
            const float bias0 = rel_table[(size_t)(d0 + MAXD) * 16 + head];
            const float bias1 = rel_table[(size_t)(d1 + MAXD) * 16 + head];
            s0[j] = (k0i < S) ? c0[j] * 0.125f + bias0 : -1e30f;
            s1[j] = (k1i < S) ? c1[j] * 0.125f + bias1 : -1e30f;
        }

        // ---- row max across the 16 lanes of this half ----
        float cmax[8];
#pragma unroll
        for (int j = 0; j < 8; ++j) cmax[j] = fmaxf(s0[j], s1[j]);
#pragma unroll
        for (int off = 1; off < 16; off <<= 1)
#pragma unroll
            for (int j = 0; j < 8; ++j) cmax[j] = fmaxf(cmax[j], __shfl_xor(cmax[j], off, 32));

        // ---- online softmax update; stage P into LDS (f16) ----
#pragma unroll
        for (int j = 0; j < 8; ++j) {
            const float mnew  = fmaxf(mrow[j], cmax[j]);
            const float scale = __expf(mrow[j] - mnew);
            const float p0 = __expf(s0[j] - mnew);
            const float p1 = __expf(s1[j] - mnew);
            mrow[j] = mnew;
            accl[j] *= scale;
#pragma unroll
            for (int nt = 0; nt < 4; ++nt) acc[nt][j] *= scale;
            s_p[wave][8 * h + j][m]      = (_Float16)p0;
            s_p[wave][8 * h + j][16 + m] = (_Float16)p1;
        }

        // same-wave LDS store -> load: hardware keeps DS in order; block compiler reordering
        asm volatile("s_wait_dscnt 0" ::: "memory");

        // ---- P A-fragment (16 q x 32 chunk-local keys): 2 x b128 ----
        V16 pa;
        {
            const _Float16* src = &s_p[wave][m][0];
            LOAD_F4(pa.f4[0], src + 16 * h);
            LOAD_F4(pa.f4[1], src + 16 * h + 8);
        }

        // ---- P @ V (4 WMMAs) + P @ 1 (row sums) ----
#pragma unroll
        for (int nt = 0; nt < 4; ++nt) {
            V16 vb;
            const _Float16* src = &s_vt[nt * 16 + m][16 * h];
            LOAD_F4(vb.f4[0], src);
            LOAD_F4(vb.f4[1], src + 8);
            acc[nt] = WMMA_F16(pa.v, vb.v, acc[nt]);
        }
        accl = WMMA_F16(pa.v, ones.v, accl);
    }

    // ---- normalize + store: ao[tok, head*64 + d] ----
#pragma unroll
    for (int j = 0; j < 8; ++j) {
        const int qr = q0 + 8 * h + j;
        if (qr < S) {
            const float invl = 1.0f / accl[j];
            const size_t base = (size_t)b * S * 1024 + (size_t)qr * 1024 + (size_t)head * 64;
#pragma unroll
            for (int nt = 0; nt < 4; ++nt)
                ao[base + nt * 16 + m] = (_Float16)(acc[nt][j] * invl);
        }
    }
}

// ---------------------------------------------------------------------------
extern "C" void kernel_launch(void* const* d_in, const int* in_sizes, int n_in,
                              void* d_out, int out_size, void* d_ws, size_t ws_size,
                              hipStream_t stream) {
    const float* query     = (const float*)d_in[0];
    const float* key       = (const float*)d_in[1];
    const float* value     = (const float*)d_in[2];
    const float* Wq        = (const float*)d_in[3];
    const float* Wk        = (const float*)d_in[4];
    const float* Wv        = (const float*)d_in[5];
    const float* Wo        = (const float*)d_in[6];
    const float* rel_table = (const float*)d_in[7];

    constexpr int B = 32, S = 484, E = 1024, H = 1024, KV = 256;
    constexpr int M = B * S;  // 15488 = 121 * 128

    _Float16* qp = (_Float16*)d_ws;
    _Float16* kp = qp + (size_t)M * H;
    _Float16* vp = kp + (size_t)M * KV;
    _Float16* ao = vp + (size_t)M * KV;

    const dim3 blk(128);
    // projections (fp32 in, f16 out)
    gemm_wmma<false, false><<<dim3(M / 128, H / 64),  blk, 0, stream>>>(query, Wq, qp, M, H,  E);
    gemm_wmma<false, false><<<dim3(M / 128, KV / 64), blk, 0, stream>>>(key,   Wk, kp, M, KV, E);
    gemm_wmma<false, false><<<dim3(M / 128, KV / 64), blk, 0, stream>>>(value, Wv, vp, M, KV, E);
    // attention: 32 batches * 16 heads * 8 q-tiles of 64 rows
    attn_wmma<<<dim3(B * 16 * 8), blk, 0, stream>>>(qp, kp, vp, rel_table, ao);
    // output projection (f16 in, fp32 out)
    gemm_wmma<true, true><<<dim3(M / 128, H / 64), blk, 0, stream>>>(ao, Wo, (float*)d_out, M, H, E);
}
